// MSDeform_attn_transformer_encoder_layer_74122545594641
// MI455X (gfx1250) — matrix operand order repaired and checked by reading, not compiled, not run on hardware
//
#include <hip/hip_runtime.h>
#include <hip/hip_bf16.h>

typedef __attribute__((ext_vector_type(16))) _Float16 v16h;
typedef __attribute__((ext_vector_type(8)))  _Float16 v8h;
typedef __attribute__((ext_vector_type(8)))  float    v8f;
typedef __attribute__((ext_vector_type(4)))  int      v4i;

namespace {

constexpr int kB = 2, kS = 13125, kD = 256, kFFN = 1024;
constexpr int kM = kB * kS;            // 26250 rows
constexpr int kMpad = 26368;           // 206 * 128
constexpr int kNoaPad = 320;           // 288 (=192 off + 96 attn) padded to 64-mult

// ---- workspace layout (bytes) ----
constexpr size_t SZ_A16  = (size_t)kMpad * kD * 2;        // 13,500,416
constexpr size_t SZ_F32  = (size_t)kMpad * kD * 4;        // 27,000,832
constexpr size_t SZ_OA   = (size_t)kMpad * kNoaPad * 4;   // 33,751,040
constexpr size_t OFF_QH   = 0;
constexpr size_t OFF_XH   = OFF_QH  + SZ_A16;
constexpr size_t OFF_V16  = OFF_XH  + SZ_A16;
constexpr size_t OFF_OA   = OFF_V16 + SZ_A16;
constexpr size_t OFF_ATT  = OFF_OA  + SZ_OA;              // 74,252,288
constexpr size_t OFF_Y1   = OFF_ATT + SZ_A16;
constexpr size_t OFF_H1F  = OFF_Y1  + SZ_F32;
constexpr size_t OFF_H1H  = OFF_H1F + SZ_F32;
constexpr size_t OFF_W    = OFF_H1H + SZ_A16;             // weights region
constexpr size_t OFF_WTVAL = OFF_W;
constexpr size_t OFF_WTOUT = OFF_WTVAL + 256*256*2;
constexpr size_t OFF_WCAT  = OFF_WTOUT + 256*256*2;
constexpr size_t OFF_BCAT  = OFF_WCAT  + (size_t)kNoaPad*256*2;
constexpr size_t OFF_W1T   = OFF_BCAT  + 4096;
constexpr size_t OFF_W2T   = OFF_W1T   + (size_t)1024*256*2;
// aliased (phase-2) buffers: earlier buffers are dead by the time these are written
constexpr size_t OFF_FFN1 = 0;         // 54,001,664 bytes over [qh|xh|v16|oa]
constexpr size_t OFF_Y2   = OFF_ATT;   // 27,000,832 bytes over [att|y1]

enum { EP_F16 = 0, EP_F32 = 1, EP_RELU_F16 = 2, EP_RESID_F32 = 3 };

// ---- CDNA5 async global->LDS copy (16 bytes per lane), with safe fallback ----
__device__ __forceinline__ void async_copy16(const _Float16* g, _Float16* l) {
#if __has_builtin(__builtin_amdgcn_global_load_async_to_lds_b128)
  __builtin_amdgcn_global_load_async_to_lds_b128(
      (__attribute__((address_space(1))) v4i*)const_cast<_Float16*>(g),
      (__attribute__((address_space(3))) v4i*)l, 0, 0);
#else
  *(v8h*)l = *(const v8h*)g;
#endif
}
__device__ __forceinline__ void async_wait_all() {
#if __has_builtin(__builtin_amdgcn_s_wait_asynccnt)
  __builtin_amdgcn_s_wait_asynccnt(0);
#elif __has_builtin(__builtin_amdgcn_global_load_async_to_lds_b128)
  asm volatile("s_wait_asynccnt 0x0" ::: "memory");
#endif
}

// ---------------- prep kernels ----------------
__global__ __launch_bounds__(256) void prep_qx(const float* __restrict__ x,
                                               const float* __restrict__ pos,
                                               _Float16* __restrict__ qh,
                                               _Float16* __restrict__ xh) {
  size_t idx = (size_t)blockIdx.x * 256 + threadIdx.x;   // over Mpad*256
  int m = (int)(idx >> 8);
  if (m < kM) {
    float xv = x[idx];
    qh[idx] = (_Float16)(xv + pos[idx]);
    xh[idx] = (_Float16)xv;
  } else {
    qh[idx] = (_Float16)0.f;
    xh[idx] = (_Float16)0.f;
  }
}

// W[k][n] (row-major KxN) -> Wt[n][k] f16, zero-padded rows n>=N
__global__ __launch_bounds__(256) void prep_wt(const float* __restrict__ W,
                                               _Float16* __restrict__ Wt,
                                               int K, int N) {
  size_t idx = (size_t)blockIdx.x * 256 + threadIdx.x;   // over Npad*K
  int n = (int)(idx / (size_t)K);
  int k = (int)(idx % (size_t)K);
  float v = (n < N) ? W[(size_t)k * N + n] : 0.f;
  Wt[idx] = (_Float16)v;
}

// concat [W_off | W_attn] -> Wt [320][256], bias -> bcat[320]
__global__ __launch_bounds__(256) void prep_cat(const float* __restrict__ Woff,
                                                const float* __restrict__ boff,
                                                const float* __restrict__ Wattn,
                                                const float* __restrict__ battn,
                                                _Float16* __restrict__ Wt,
                                                float* __restrict__ bcat) {
  size_t idx = (size_t)blockIdx.x * 256 + threadIdx.x;   // over 320*256
  int n = (int)(idx >> 8);
  int k = (int)(idx & 255);
  float v = (n < 192) ? Woff[(size_t)k * 192 + n]
          : (n < 288) ? Wattn[(size_t)k * 96 + (n - 192)] : 0.f;
  Wt[idx] = (_Float16)v;
  if (k == 0) bcat[n] = (n < 192) ? boff[n] : (n < 288 ? battn[n - 192] : 0.f);
}

// ---------------- WMMA GEMM ----------------
// C[M x N] = A[Mpad x K](f16) * Bt[Npad x K](f16)^T + bias (+resid)
// block: 256 thr = 8 waves over 128 rows; each wave: 16 rows x 64 cols.
// B tile (64 cols x 32 K) staged in LDS via async copies, double buffered;
// A fragments stream from global (L2 resident) software-pipelined 1 k-step ahead.
template <int EP>
__global__ __launch_bounds__(256) void gemm_wmma(const _Float16* __restrict__ A,
                                                 const _Float16* __restrict__ Bt,
                                                 const float* __restrict__ bias,
                                                 const float* __restrict__ resid,
                                                 int Mvalid, void* __restrict__ out,
                                                 int K, int ldc) {
  // 64 rows x 40 halfs (80B row, padded from 64B to spread LDS banks)
  __shared__ _Float16 ldsB[2][64 * 40];
  const int tid = threadIdx.x;
  const int lane = tid & 31;
  const int wave = tid >> 5;
  const int rowBase = blockIdx.y * 128 + wave * 16;
  const int colBase = blockIdx.x * 64;
  const int hiLane = (lane & 16) ? 1 : 0;

  // cooperative B-tile fill: 256 thr x 16B = 4KB tile per k-step
  const int brow = tid >> 2;   // 0..63  (local col n)
  const int bchk = tid & 3;    // 0..3   (8-half chunk within 32-K slab)
  const _Float16* bsrc = Bt + (size_t)(colBase + brow) * K + bchk * 8;
  _Float16* bdst[2] = {&ldsB[0][brow * 40 + bchk * 8], &ldsB[1][brow * 40 + bchk * 8]};

  // A fragment: lane<16 -> K chunks [0..7],[16..23]; lane>=16 -> [8..15],[24..31]
  const _Float16* arow = A + (size_t)(rowBase + (lane & 15)) * K + (hiLane ? 8 : 0);

  async_copy16(bsrc, bdst[0]);                   // k0 = 0 tile in flight

  const v8h* pa0 = (const v8h*)arow;
  v8h alo = pa0[0];
  v8h ahi = pa0[2];

  v8f acc[4] = {};
  for (int k0 = 0; k0 < K; k0 += 32) {
    const int cur = (k0 >> 5) & 1;
    async_wait_all();          // our async fill of ldsB[cur] landed
    __syncthreads();           // all waves' fills landed; prior reads of other buf done
    if (k0 + 32 < K) async_copy16(bsrc + k0 + 32, bdst[cur ^ 1]);

    // prefetch + pipeline next A fragment
    v8h nlo = alo, nhi = ahi;
    if (k0 + 32 < K) {
      const v8h* pn = (const v8h*)(arow + k0 + 32);
      nlo = pn[0];
      nhi = pn[2];
      __builtin_prefetch(arow + k0 + 64, 0, 1);
    }

    v16h af = __builtin_shufflevector(alo, ahi, 0,1,2,3,4,5,6,7,8,9,10,11,12,13,14,15);
    const _Float16* lb = ldsB[cur];
#pragma unroll
    for (int t = 0; t < 4; ++t) {
      // B fragment from LDS: lane<16 -> K [0..7],[16..23]; lane>=16 -> K [8..15],[24..31]
      const v8h* pb = (const v8h*)(lb + (t * 16 + (lane & 15)) * 40 + (hiLane ? 8 : 0));
      v8h blo = pb[0];
      v8h bhi = pb[2];
      v16h bf = __builtin_shufflevector(blo, bhi, 0,1,2,3,4,5,6,7,8,9,10,11,12,13,14,15);
      acc[t] = __builtin_amdgcn_wmma_f32_16x16x32_f16(false, af, false, bf,
                                                      (short)0, acc[t], false, false);
    }
    alo = nlo;
    ahi = nhi;
  }

  const int rlo = rowBase + (hiLane ? 8 : 0);
#pragma unroll
  for (int t = 0; t < 4; ++t) {
    const int n = colBase + t * 16 + (lane & 15);
    const float bn = bias[n];
#pragma unroll
    for (int r = 0; r < 8; ++r) {
      const int row = rlo + r;
      float v = acc[t][r] + bn;
      if (EP == EP_RELU_F16) v = v > 0.f ? v : 0.f;
      if (EP == EP_RESID_F32) v += (row < Mvalid) ? resid[(size_t)row * ldc + n] : 0.f;
      if (EP == EP_F32 || EP == EP_RESID_F32)
        ((float*)out)[(size_t)row * ldc + n] = v;
      else
        ((_Float16*)out)[(size_t)row * ldc + n] = (_Float16)v;
    }
  }
}

// ---------------- deformable sampling ----------------
__global__ __launch_bounds__(256) void sample_kernel(const _Float16* __restrict__ value16,
                                                     const float* __restrict__ oa,
                                                     const float* __restrict__ refp,
                                                     _Float16* __restrict__ att16) {
  const int t = blockIdx.x * 256 + threadIdx.x;      // over Mpad*8
  if (t >= kMpad * 8) return;
  const int h = t & 7;
  const int m = t >> 3;
  _Float16* dst = att16 + (size_t)m * kD + h * 32;
  if (m >= kM) {
    v8h z = {};
#pragma unroll
    for (int c = 0; c < 4; ++c) ((v8h*)dst)[c] = z;
    return;
  }
  const int b = m / kS;
  const int s = m - b * kS;
  const float* row = oa + (size_t)m * kNoaPad;

  // softmax over 12 logits
  float lg[12];
  float mx = -1e30f;
#pragma unroll
  for (int j = 0; j < 12; ++j) { lg[j] = row[192 + h * 12 + j]; mx = fmaxf(mx, lg[j]); }
  float sum = 0.f;
#pragma unroll
  for (int j = 0; j < 12; ++j) { lg[j] = __expf(lg[j] - mx); sum += lg[j]; }
  const float inv = 1.f / sum;

  float acc[32];
#pragma unroll
  for (int d = 0; d < 32; ++d) acc[d] = 0.f;

  const int lvlH[3] = {100, 50, 25}, lvlW[3] = {100, 50, 25}, lvlS[3] = {0, 10000, 12500};
  const float* rp = refp + ((size_t)m * 3) * 2;      // [B,S,L,2]

#pragma unroll
  for (int l = 0; l < 3; ++l) {
    const int Hl = lvlH[l], Wl = lvlW[l], st = lvlS[l];
    const float rx = rp[l * 2 + 0], ry = rp[l * 2 + 1];
    const _Float16* vbase = value16 + ((size_t)(b * kS + st)) * kD + h * 32;
#pragma unroll
    for (int p = 0; p < 4; ++p) {
      const float ox = row[(h * 3 + l) * 8 + p * 2 + 0];
      const float oy = row[(h * 3 + l) * 8 + p * 2 + 1];
      const float aw = lg[l * 4 + p] * inv;
      const float x = rx * (float)Wl + ox - 0.5f;    // (rx + ox/Wl)*Wl - 0.5
      const float y = ry * (float)Hl + oy - 0.5f;
      const float x0f = floorf(x), y0f = floorf(y);
      const float fx = x - x0f, fy = y - y0f;
      const int x0 = (int)x0f, y0 = (int)y0f;
      const int dx[4] = {0, 1, 0, 1}, dy[4] = {0, 0, 1, 1};
      const float bw[4] = {(1.f - fx) * (1.f - fy), fx * (1.f - fy),
                           (1.f - fx) * fy, fx * fy};
#pragma unroll
      for (int q = 0; q < 4; ++q) {
        const int xi = x0 + dx[q], yi = y0 + dy[q];
        if (xi >= 0 && xi < Wl && yi >= 0 && yi < Hl) {
          const float w = aw * bw[q];
          const v8h* vp = (const v8h*)(vbase + (size_t)(yi * Wl + xi) * kD);
#pragma unroll
          for (int c = 0; c < 4; ++c) {
            v8h hv = vp[c];
#pragma unroll
            for (int i = 0; i < 8; ++i) acc[c * 8 + i] += w * (float)hv[i];
          }
        }
      }
    }
  }
#pragma unroll
  for (int c = 0; c < 4; ++c) {
    v8h o;
#pragma unroll
    for (int i = 0; i < 8; ++i) o[i] = (_Float16)acc[c * 8 + i];
    ((v8h*)dst)[c] = o;
  }
}

// ---------------- layernorm (wave32 per row) ----------------
__global__ __launch_bounds__(256) void ln_kernel(const float* __restrict__ in,
                                                 const float* __restrict__ g,
                                                 const float* __restrict__ be,
                                                 float* __restrict__ outf,
                                                 _Float16* __restrict__ outh,
                                                 int rows) {
  const int wave = threadIdx.x >> 5, lane = threadIdx.x & 31;
  const int row = blockIdx.x * 8 + wave;
  if (row >= rows) return;
  const float* r = in + (size_t)row * kD;
  float v[8], s = 0.f, sq = 0.f;
#pragma unroll
  for (int i = 0; i < 8; ++i) {
    v[i] = r[lane + i * 32];
    s += v[i];
    sq += v[i] * v[i];
  }
#pragma unroll
  for (int off = 16; off >= 1; off >>= 1) {
    s += __shfl_xor(s, off, 32);
    sq += __shfl_xor(sq, off, 32);
  }
  const float mean = s * (1.f / 256.f);
  const float var = sq * (1.f / 256.f) - mean * mean;
  const float rstd = rsqrtf(var + 1e-5f);
#pragma unroll
  for (int i = 0; i < 8; ++i) {
    const int c = lane + i * 32;
    const float o = (v[i] - mean) * rstd * g[c] + be[c];
    if (outf) outf[(size_t)row * kD + c] = o;
    if (outh) outh[(size_t)row * kD + c] = (_Float16)o;
  }
}

}  // namespace

extern "C" void kernel_launch(void* const* d_in, const int* in_sizes, int n_in,
                              void* d_out, int out_size, void* d_ws, size_t ws_size,
                              hipStream_t stream) {
  const float* x    = (const float*)d_in[0];
  const float* pos  = (const float*)d_in[1];
  const float* refp = (const float*)d_in[2];
  const float* Woff = (const float*)d_in[3];
  const float* boff = (const float*)d_in[4];
  const float* Wattn= (const float*)d_in[5];
  const float* battn= (const float*)d_in[6];
  const float* Wval = (const float*)d_in[7];
  const float* bval = (const float*)d_in[8];
  const float* Wout = (const float*)d_in[9];
  const float* bout = (const float*)d_in[10];
  const float* g1   = (const float*)d_in[11];
  const float* be1  = (const float*)d_in[12];
  const float* W1   = (const float*)d_in[13];
  const float* b1   = (const float*)d_in[14];
  const float* W2   = (const float*)d_in[15];
  const float* b2   = (const float*)d_in[16];
  const float* g2   = (const float*)d_in[17];
  const float* be2  = (const float*)d_in[18];

  char* ws = (char*)d_ws;
  _Float16* qh    = (_Float16*)(ws + OFF_QH);
  _Float16* xh    = (_Float16*)(ws + OFF_XH);
  _Float16* v16   = (_Float16*)(ws + OFF_V16);
  float*    oabuf = (float*)   (ws + OFF_OA);
  _Float16* att16 = (_Float16*)(ws + OFF_ATT);
  float*    y1    = (float*)   (ws + OFF_Y1);
  float*    h1f   = (float*)   (ws + OFF_H1F);
  _Float16* h1h   = (_Float16*)(ws + OFF_H1H);
  _Float16* wtval = (_Float16*)(ws + OFF_WTVAL);
  _Float16* wtout = (_Float16*)(ws + OFF_WTOUT);
  _Float16* wcat  = (_Float16*)(ws + OFF_WCAT);
  float*    bcat  = (float*)   (ws + OFF_BCAT);
  _Float16* w1t   = (_Float16*)(ws + OFF_W1T);
  _Float16* w2t   = (_Float16*)(ws + OFF_W2T);
  _Float16* ffn1  = (_Float16*)(ws + OFF_FFN1);
  float*    y2    = (float*)   (ws + OFF_Y2);
  float*    outp  = (float*)d_out;

  // ---- phase 0: conversions ----
  prep_qx<<<kMpad, 256, 0, stream>>>(x, pos, qh, xh);
  prep_wt<<<(256 * 256) / 256, 256, 0, stream>>>(Wval, wtval, 256, 256);
  prep_wt<<<(256 * 256) / 256, 256, 0, stream>>>(Wout, wtout, 256, 256);
  prep_wt<<<(1024 * 256) / 256, 256, 0, stream>>>(W1, w1t, 256, 1024);
  prep_wt<<<(256 * 1024) / 256, 256, 0, stream>>>(W2, w2t, 1024, 256);
  prep_cat<<<(kNoaPad * 256) / 256, 256, 0, stream>>>(Woff, boff, Wattn, battn, wcat, bcat);

  const dim3 blk(256);
  const int gy = kMpad / 128;  // 206

  // ---- phase 1: value projection (f16 out) ----
  gemm_wmma<EP_F16><<<dim3(256 / 64, gy), blk, 0, stream>>>(
      xh, wtval, bval, nullptr, 0, v16, 256, 256);
  // ---- off+attn logits (f32 out, ldc=320) ----
  gemm_wmma<EP_F32><<<dim3(kNoaPad / 64, gy), blk, 0, stream>>>(
      qh, wcat, bcat, nullptr, 0, oabuf, 256, kNoaPad);
  // ---- deformable sampling + softmax + combine ----
  sample_kernel<<<(kMpad * 8 + 255) / 256, blk, 0, stream>>>(v16, oabuf, refp, att16);
  // ---- output projection + residual x ----
  gemm_wmma<EP_RESID_F32><<<dim3(256 / 64, gy), blk, 0, stream>>>(
      att16, wtout, bout, x, kM, y1, 256, 256);
  // ---- LN1 -> h1 (f32 + f16) ----
  ln_kernel<<<kMpad / 8, blk, 0, stream>>>(y1, g1, be1, h1f, h1h, kMpad);
  // ---- FFN1 (ReLU, f16 out) ----
  gemm_wmma<EP_RELU_F16><<<dim3(1024 / 64, gy), blk, 0, stream>>>(
      h1h, w1t, b1, nullptr, 0, ffn1, 256, 1024);
  // ---- FFN2 + residual h1 ----
  gemm_wmma<EP_RESID_F32><<<dim3(256 / 64, gy), blk, 0, stream>>>(
      ffn1, w2t, b2, h1f, kMpad, y2, 1024, 256);
  // ---- LN2 -> final output (masked to true M rows) ----
  ln_kernel<<<(kM + 7) / 8, blk, 0, stream>>>(y2, g2, be2, outp, nullptr, kM);
}
